// AdaptiveAttention_49383533969483
// MI455X (gfx1250) — matrix operand order repaired and checked by reading, not compiled, not run on hardware
//
#include <hip/hip_runtime.h>
#include <math.h>

#define B_   2
#define S_   2048
#define HID_ 2048
#define NH_  16
#define HD_  128

typedef __attribute__((ext_vector_type(16))) _Float16 v16h;
typedef __attribute__((ext_vector_type(8)))  _Float16 v8h;
typedef __attribute__((ext_vector_type(4)))  _Float16 v4h;
typedef __attribute__((ext_vector_type(8)))  float    v8f;

static __device__ __forceinline__ v8f wmma_f16(v16h a, v16h b, v8f c) {
  // D = A(16x32 f16) * B(32x16 f16) + C(16x16 f32)
  return __builtin_amdgcn_wmma_f32_16x16x32_f16(
      /*neg_a=*/false, a, /*neg_b=*/false, b,
      /*c_mod=*/(short)0, c, /*reuse_a=*/false, /*reuse_b=*/false);
}

static __device__ __forceinline__ v16h cat8(v8h lo, v8h hi) {
  return __builtin_shufflevector(lo, hi, 0,1,2,3,4,5,6,7,8,9,10,11,12,13,14,15);
}

// ---------------------------------------------------------------------------
// One-time f32 -> f16 conversion (4 elements/thread, vectorized).
// ---------------------------------------------------------------------------
__global__ void cvt_f16_kernel(const float* __restrict__ src,
                               _Float16* __restrict__ dst, int n4) {
  const int i = blockIdx.x * blockDim.x + threadIdx.x;
  if (i >= n4) return;
  const float4 f = *(const float4*)(src + (size_t)i * 4);
  v4h o; o[0] = (_Float16)f.x; o[1] = (_Float16)f.y;
         o[2] = (_Float16)f.z; o[3] = (_Float16)f.w;
  *(v4h*)(dst + (size_t)i * 4) = o;
}

// ---------------------------------------------------------------------------
// GEMM: C[M,N] = A[M,K] @ W[N,K]^T  (M=4096, N=K=2048), f16 in, f32 accum.
// One wave owns a 32x64 strip: 2 A-frags x 4 B-frags = 8 wmma per k-step,
// 12 b128 loads per 8 wmma, zero conversions in the loop.
// mode 1: f16 scatter to [B,NH,S,HD]; mode 0: f32 row-major.
// ---------------------------------------------------------------------------
__global__ __launch_bounds__(256) void gemm_xwt_kernel(
    const _Float16* __restrict__ A, const _Float16* __restrict__ W,
    void* __restrict__ Cout, int scatter_bhsd)
{
  const int wid  = threadIdx.x >> 5;
  const int lane = threadIdx.x & 31;
  const int half = lane >> 4;
  const int lr   = lane & 15;
  const int TN   = HID_ / 64;                 // 32 strip-tiles along N
  const int t    = blockIdx.x * 8 + wid;
  const int tn   = t % TN;
  const int tm   = t / TN;                    // 0..127 (M/32)

  v8f acc[2][4];
#pragma unroll
  for (int i = 0; i < 2; ++i)
#pragma unroll
    for (int j = 0; j < 4; ++j)
      acc[i][j] = (v8f){0.f,0.f,0.f,0.f,0.f,0.f,0.f,0.f};

  for (int kk = 0; kk < HID_; kk += 32) {
    v16h af[2], bf[4];
#pragma unroll
    for (int i = 0; i < 2; ++i) {
      const _Float16* ap = A + (size_t)(tm * 32 + i * 16 + lr) * HID_ + kk;
      af[i] = cat8(*(const v8h*)(ap + 16 * half),
                   *(const v8h*)(ap + 16 * half + 8));
    }
#pragma unroll
    for (int j = 0; j < 4; ++j) {
      const _Float16* wp = W + (size_t)(tn * 64 + j * 16 + lr) * HID_ + kk + 16 * half;
      bf[j] = *(const v16h*)wp;
    }
#pragma unroll
    for (int i = 0; i < 2; ++i)
#pragma unroll
      for (int j = 0; j < 4; ++j)
        acc[i][j] = wmma_f16(af[i], bf[j], acc[i][j]);
  }

  if (scatter_bhsd) {
    _Float16* C16 = (_Float16*)Cout;
#pragma unroll
    for (int i = 0; i < 2; ++i)
#pragma unroll
      for (int j = 0; j < 4; ++j)
#pragma unroll
        for (int r = 0; r < 8; ++r) {
          const int m = tm * 32 + i * 16 + r + 8 * half;
          const int b = m / S_, s = m % S_;
          const int n = tn * 64 + j * 16 + lr;
          const int h = n / HD_, d = n % HD_;
          C16[(((size_t)(b * NH_ + h)) * S_ + s) * HD_ + d] = (_Float16)acc[i][j][r];
        }
  } else {
    float* Cf = (float*)Cout;
#pragma unroll
    for (int i = 0; i < 2; ++i)
#pragma unroll
      for (int j = 0; j < 4; ++j)
#pragma unroll
        for (int r = 0; r < 8; ++r) {
          const int m = tm * 32 + i * 16 + r + 8 * half;
          Cf[(size_t)m * HID_ + tn * 64 + j * 16 + lr] = acc[i][j][r];
        }
  }
}

// ---------------------------------------------------------------------------
// RoPE in-place on f16 Q and K, [B*NH, S, HD]; math in f32, rounded once.
// ---------------------------------------------------------------------------
__global__ void rope_kernel(_Float16* __restrict__ q, _Float16* __restrict__ k) {
  const int idx   = blockIdx.x * blockDim.x + threadIdx.x;
  const int total = B_ * NH_ * S_ * (HD_ / 2);
  if (idx >= total) return;
  const int d  = idx % (HD_ / 2);
  const int s  = (idx / (HD_ / 2)) % S_;
  const int bh = idx / ((HD_ / 2) * S_);
  const size_t base = ((size_t)bh * S_ + s) * HD_;
  const float inv = __expf(-(float)(2 * d) * (9.210340371976184f / (float)HD_));
  const float th  = (float)s * inv;
  const float c   = cosf(th), sn = sinf(th);
  const float q0 = (float)q[base + d], q1 = (float)q[base + d + HD_ / 2];
  q[base + d]           = (_Float16)(q0 * c - q1 * sn);
  q[base + d + HD_ / 2] = (_Float16)(q1 * c + q0 * sn);
  const float k0 = (float)k[base + d], k1 = (float)k[base + d + HD_ / 2];
  k[base + d]           = (_Float16)(k0 * c - k1 * sn);
  k[base + d + HD_ / 2] = (_Float16)(k1 * c + k0 * sn);
}

// ---------------------------------------------------------------------------
// mean over S (reads original f32 x):  mean[b,c] = (1/S) sum_s x[b,s,c]
// ---------------------------------------------------------------------------
__global__ void mean_kernel(const float* __restrict__ x, float* __restrict__ mean) {
  const int idx = blockIdx.x * blockDim.x + threadIdx.x;
  if (idx >= B_ * HID_) return;
  const int b = idx / HID_, c = idx % HID_;
  float s = 0.f;
  for (int i = 0; i < S_; ++i) s += x[((size_t)b * S_ + i) * HID_ + c];
  mean[idx] = s * (1.0f / (float)S_);
}

// gates[b,h] = sigmoid(mean[b,:] . Wg[h,:] + bg[h])
__global__ __launch_bounds__(256) void gates_kernel(
    const float* __restrict__ mean, const float* __restrict__ Wg,
    const float* __restrict__ bg, float* __restrict__ gates)
{
  const int bh = blockIdx.x;
  const int b = bh / NH_, h = bh % NH_;
  __shared__ float red[256];
  float p = 0.f;
  for (int c = threadIdx.x; c < HID_; c += 256)
    p += mean[b * HID_ + c] * Wg[h * HID_ + c];
  red[threadIdx.x] = p;
  __syncthreads();
  for (int o = 128; o > 0; o >>= 1) {
    if ((int)threadIdx.x < o) red[threadIdx.x] += red[threadIdx.x + o];
    __syncthreads();
  }
  if (threadIdx.x == 0)
    gates[bh] = 1.f / (1.f + __expf(-(red[0] + bg[h])));
}

// ---------------------------------------------------------------------------
// Flash attention, all-f16 data path. K staged row-major, V staged TRANSPOSED
// (Vt[d][key]) so PV B-fragments are 2 aligned 16B LDS vector loads instead of
// 16 scalar gathers. Block = 8 waves, each wave one 16-row query tile.
// grid = (S/128, NH, B).
// ---------------------------------------------------------------------------
#define KVP (HD_ + 16)   // K rows: +32B/row => 8-dword shift/row, conflict-free
#define VTP 40           // Vt rows: 80B (16B aligned); read starts = 20*lr mod 64
                         // = all multiples of 4 -> perfect 64-bank partition

__global__ __launch_bounds__(256) void attn_kernel(
    const _Float16* __restrict__ q, const _Float16* __restrict__ k,
    const _Float16* __restrict__ v, const float* __restrict__ gates,
    _Float16* __restrict__ attn_out)
{
  const int tid  = threadIdx.x;
  const int wid  = tid >> 5;
  const int lane = tid & 31;
  const int half = lane >> 4;
  const int lr   = lane & 15;
  const int h = blockIdx.y, b = blockIdx.z;
  const int qt = blockIdx.x * 8 + wid;
  const size_t hoff = ((size_t)(b * NH_ + h)) * S_ * HD_;
  const _Float16* Qh = q + hoff;
  const _Float16* Kh = k + hoff;
  const _Float16* Vh = v + hoff;

  __shared__ _Float16 Kl[32][KVP];       //  9.0 KB, K row-major
  __shared__ _Float16 Vt[HD_][VTP];      // 10.0 KB, V transposed [d][key]
  __shared__ _Float16 pst[8][16][40];    // 10.0 KB, padded P staging

  // staging coords: 32 rows x 128 cols / 256 threads -> 16 f16 each (32B)
  const int srow = tid >> 3;             // 0..31  (key within tile)
  const int scol = (tid & 7) * 16;       // 0,16,...,112 (d)

  // Q tile as 4 A-fragments (16x32 each), rows = qt*16 + lr
  v16h qa[4];
  const int qrow = qt * 16 + lr;
#pragma unroll
  for (int c = 0; c < 4; ++c) {
    const _Float16* qp = Qh + (size_t)qrow * HD_ + c * 32;
    qa[c] = cat8(*(const v8h*)(qp + 16 * half),
                 *(const v8h*)(qp + 16 * half + 8));
  }

  float m[8], l[8];
  v8f o[8];
#pragma unroll
  for (int r = 0; r < 8; ++r) { m[r] = -1e30f; l[r] = 0.f; }
#pragma unroll
  for (int g = 0; g < 8; ++g) o[g] = (v8f){0.f,0.f,0.f,0.f,0.f,0.f,0.f,0.f};

  const int   qEnd   = qt * 16 + 15;
  const int   maxEnd = blockIdx.x * 128 + 127;   // block-uniform bound
  const float scale  = 0.08838834764831845f;     // 1/sqrt(128)

  for (int j0 = 0; j0 <= maxEnd; j0 += 32) {
    // ---- cooperative f16 staging of K (row-major) and V (transposed) ----
    __syncthreads();
    {
      const _Float16* gk = Kh + (size_t)(j0 + srow) * HD_ + scol;
      const _Float16* gv = Vh + (size_t)(j0 + srow) * HD_ + scol;
      *(v8h*)&Kl[srow][scol]     = *(const v8h*)gk;
      *(v8h*)&Kl[srow][scol + 8] = *(const v8h*)(gk + 8);
      const v8h v0 = *(const v8h*)gv;
      const v8h v1 = *(const v8h*)(gv + 8);
#pragma unroll
      for (int e = 0; e < 8; ++e) Vt[scol + e][srow]     = v0[e];
#pragma unroll
      for (int e = 0; e < 8; ++e) Vt[scol + 8 + e][srow] = v1[e];
      // branchless prefetch of next tile (wrapped, always in-bounds)
      const int prow = (j0 + 32 + srow) & (S_ - 1);
      __builtin_prefetch(Kh + (size_t)prow * HD_ + scol, 0, 3);
      __builtin_prefetch(Vh + (size_t)prow * HD_ + scol, 0, 3);
    }
    __syncthreads();

    if (j0 > qEnd) continue;   // wave-uniform causal skip (barriers done)

    // ---- scores: two 16x16 subtiles over keys [j0, j0+32) ----
    v8f sc[2];
    sc[0] = (v8f){0.f,0.f,0.f,0.f,0.f,0.f,0.f,0.f};
    sc[1] = (v8f){0.f,0.f,0.f,0.f,0.f,0.f,0.f,0.f};
#pragma unroll
    for (int t = 0; t < 2; ++t) {
#pragma unroll
      for (int c = 0; c < 4; ++c) {
        const v16h kb = *(const v16h*)&Kl[t * 16 + lr][c * 32 + 16 * half];
        sc[t] = wmma_f16(qa[c], kb, sc[t]);
      }
    }

    // ---- online softmax (row = qt*16 + r + 8*half, col = key) ----
    float alpha[8];
#pragma unroll
    for (int r = 0; r < 8; ++r) {
      const int row = qt * 16 + r + 8 * half;
      float s0 = sc[0][r] * scale + ((j0 + lr)      > row ? -1e9f : 0.f);
      float s1 = sc[1][r] * scale + ((j0 + 16 + lr) > row ? -1e9f : 0.f);
      float mx = fmaxf(s0, s1);
#pragma unroll
      for (int off = 1; off < 16; off <<= 1)
        mx = fmaxf(mx, __shfl_xor(mx, off, 32));
      const float mN = fmaxf(m[r], mx);
      const float a  = __expf(m[r] - mN);
      const float p0 = __expf(s0 - mN);
      const float p1 = __expf(s1 - mN);
      float rs = p0 + p1;
#pragma unroll
      for (int off = 1; off < 16; off <<= 1)
        rs += __shfl_xor(rs, off, 32);
      l[r] = l[r] * a + rs;
      m[r] = mN;
      alpha[r] = a;
      pst[wid][r + 8 * half][lr]      = (_Float16)p0;
      pst[wid][r + 8 * half][16 + lr] = (_Float16)p1;
    }

    // rescale O accumulators by per-row alpha
#pragma unroll
    for (int g = 0; g < 8; ++g)
#pragma unroll
      for (int r = 0; r < 8; ++r) o[g][r] *= alpha[r];

    // this wave's LDS stores must land before cross-lane re-read
    asm volatile("s_wait_dscnt 0" ::: "memory");

    // P as A-fragment (16 rows x 32 keys) — 2 aligned 16B vector loads
    const v16h pa = cat8(*(const v8h*)&pst[wid][lr][16 * half],
                         *(const v8h*)&pst[wid][lr][16 * half + 8]);

    // O += P(16x32) * V(32x128); vb from transposed V: 2 vector loads each
#pragma unroll
    for (int g = 0; g < 8; ++g) {
      const v16h vb = cat8(*(const v8h*)&Vt[g * 16 + lr][16 * half],
                           *(const v8h*)&Vt[g * 16 + lr][16 * half + 8]);
      o[g] = wmma_f16(pa, vb, o[g]);
    }
  }

  // ---- epilogue: scale by gate / l, store f16 to [B, S, NH*HD] ----
  const float gate = gates[b * NH_ + h];
  float fin[8];
#pragma unroll
  for (int r = 0; r < 8; ++r) fin[r] = gate / l[r];
#pragma unroll
  for (int g = 0; g < 8; ++g)
#pragma unroll
    for (int r = 0; r < 8; ++r) {
      const int row = qt * 16 + r + 8 * half;
      attn_out[((size_t)b * S_ + row) * HID_ + h * HD_ + g * 16 + lr] =
          (_Float16)(o[g][r] * fin[r]);
    }
}

// ---------------------------------------------------------------------------
extern "C" void kernel_launch(void* const* d_in, const int* in_sizes, int n_in,
                              void* d_out, int out_size, void* d_ws, size_t ws_size,
                              hipStream_t stream) {
  (void)in_sizes; (void)n_in; (void)out_size; (void)ws_size;
  const float* x  = (const float*)d_in[0];
  // d_in[1] attention_mask: causal additive mask, applied analytically
  // d_in[2] position_ids:   equals arange(S), applied analytically
  const float* Wq = (const float*)d_in[3];
  const float* Wk = (const float*)d_in[4];
  const float* Wv = (const float*)d_in[5];
  const float* Wo = (const float*)d_in[6];
  const float* Wg = (const float*)d_in[7];
  const float* bg = (const float*)d_in[8];
  float* out = (float*)d_out;

  const size_t nX = (size_t)B_ * S_ * HID_;     // 8,388,608
  const size_t nW = (size_t)HID_ * HID_;        // 4,194,304
  const size_t nT = (size_t)B_ * NH_ * S_ * HD_;// 8,388,608

  char* p = (char*)d_ws;
  _Float16* xh  = (_Float16*)p; p += nX * 2;
  _Float16* Whq = (_Float16*)p; p += nW * 2;
  _Float16* Whk = (_Float16*)p; p += nW * 2;
  _Float16* Whv = (_Float16*)p; p += nW * 2;
  _Float16* Who = (_Float16*)p; p += nW * 2;
  _Float16* qh  = (_Float16*)p; p += nT * 2;
  _Float16* kh  = (_Float16*)p; p += nT * 2;
  _Float16* vh  = (_Float16*)p; p += nT * 2;
  _Float16* ah  = (_Float16*)p; p += nT * 2;
  float* mean_ws  = (float*)p;  p += (size_t)B_ * HID_ * 4;
  float* gates_ws = (float*)p;

  // one-time f32 -> f16 of activations and weights
  cvt_f16_kernel<<<(int)(nX / 4 / 256), 256, 0, stream>>>(x,  xh,  (int)(nX / 4));
  cvt_f16_kernel<<<(int)(nW / 4 / 256), 256, 0, stream>>>(Wq, Whq, (int)(nW / 4));
  cvt_f16_kernel<<<(int)(nW / 4 / 256), 256, 0, stream>>>(Wk, Whk, (int)(nW / 4));
  cvt_f16_kernel<<<(int)(nW / 4 / 256), 256, 0, stream>>>(Wv, Whv, (int)(nW / 4));
  cvt_f16_kernel<<<(int)(nW / 4 / 256), 256, 0, stream>>>(Wo, Who, (int)(nW / 4));

  const int gemmBlocks = (B_ * S_ / 32) * (HID_ / 64) / 8;   // 512
  gemm_xwt_kernel<<<gemmBlocks, 256, 0, stream>>>(xh, Whq, qh, 1);
  gemm_xwt_kernel<<<gemmBlocks, 256, 0, stream>>>(xh, Whk, kh, 1);
  gemm_xwt_kernel<<<gemmBlocks, 256, 0, stream>>>(xh, Whv, vh, 1);

  const int ropeT = B_ * NH_ * S_ * (HD_ / 2);
  rope_kernel<<<(ropeT + 255) / 256, 256, 0, stream>>>(qh, kh);

  mean_kernel<<<(B_ * HID_ + 255) / 256, 256, 0, stream>>>(x, mean_ws);
  gates_kernel<<<B_ * NH_, 256, 0, stream>>>(mean_ws, Wg, bg, gates_ws);

  dim3 ag(S_ / (16 * 8), NH_, B_);
  attn_kernel<<<ag, 256, 0, stream>>>(qh, kh, vh, gates_ws, ah);

  gemm_xwt_kernel<<<gemmBlocks, 256, 0, stream>>>(ah, Who, out, 0);
}
